// StarAttn_37495064494092
// MI455X (gfx1250) — matrix operand order repaired and checked by reading, not compiled, not run on hardware
//
#include <hip/hip_runtime.h>
#include <math.h>

// StarAttn for MI455X (gfx1250, wave32).
// Algebraic refactor removes the K/V GEMMs entirely:
//   score[j,t] = A[g,t,:]·x_j + b0[g,t],   A = (1/sqrt(dh)) * q_star·Wk
//   out[t,:]   = (Wv·U[g,t,:] + bv*sumex + ex_self*v_star) / denom
//   U[g,t,:]   = sum_j ex_j * x_j
// -> two streaming passes over nodes (205 MB, ~9us @ 23.3 TB/s), GEMMs via
//    v_wmma_f32_16x16x32_f16; pass 1 stages node chunks with
//    global_load_async_to_lds_b128 (ASYNCcnt) straight into LDS.

#define HEADS    8
#define C_DIM    256
#define NUM_STAR 4
#define G_NUM    32
#define T_DIM    32                       // NUM_STAR*HEADS targets per graph
#define RSQRT_DH 0.1767766952966369f      // 1/sqrt(32)

typedef _Float16 h16;
typedef __attribute__((ext_vector_type(16))) _Float16 v16h;
typedef __attribute__((ext_vector_type(8)))  _Float16 v8h;
typedef __attribute__((ext_vector_type(8)))  float    v8f;

__device__ __forceinline__ void atomicMaxF32(float* addr, float val) {
  unsigned int* ua = (unsigned int*)addr;
  unsigned int old = __float_as_uint(*addr);
  while (__uint_as_float(old) < val) {
    unsigned int assumed = old;
    old = atomicCAS(ua, assumed, __float_as_uint(val));
    if (old == assumed) break;
  }
}

// Async copy of 16 bytes global -> LDS (gfx1250 ASYNCcnt path).
// lds_off: byte offset within the wave's LDS allocation (low 32 bits of a
// generic LDS pointer are the LDS offset per the aperture rules).
__device__ __forceinline__ void async_copy_b128(unsigned int lds_off, const void* gaddr) {
  asm volatile("global_load_async_to_lds_b128 %0, %1, off"
               :: "v"(lds_off), "v"(gaddr)
               : "memory");
}
__device__ __forceinline__ void wait_asynccnt0() {
  asm volatile("s_wait_asynccnt 0x0" ::: "memory");
}

// A-fragment (M=16 x K=32, f16) from [M][K]-row-major f16 LDS storage.
// Lane l: row = m0 + (l&15); K elements = kb + {hi*8..+7, 16+hi*8..+7}.
__device__ __forceinline__ v16h frag_a(const h16* base, int rowstride, int m0, int kb, int lane) {
  const int hi = (lane >> 4) & 1;
  const h16* p = base + (size_t)(m0 + (lane & 15)) * rowstride + kb + hi * 8;
  v8h lo = *(const v8h*)(p);
  v8h hh = *(const v8h*)(p + 16);
  v16h r;
#pragma unroll
  for (int e = 0; e < 8; ++e) { r[e] = lo[e]; r[e + 8] = hh[e]; }
  return r;
}

// Same A-fragment gather, but source rows are f32 in LDS (async-staged);
// convert to f16 at gather time.
__device__ __forceinline__ v16h frag_a_f32(const float* base, int rowstride, int m0, int kb, int lane) {
  const int hi = (lane >> 4) & 1;
  const float* p = base + (size_t)(m0 + (lane & 15)) * rowstride + kb + hi * 8;
  v16h r;
#pragma unroll
  for (int e = 0; e < 8; ++e) { r[e] = (h16)p[e]; r[e + 8] = (h16)p[e + 16]; }
  return r;
}

// B-fragment (K=32 x N=16, f16) from [N][K]-row-major ("transposed") storage.
// Lane l: col = n0 + (l&15); K elements = kb + {hi*8..+7, 16+hi*8..+7}.
__device__ __forceinline__ v16h frag_b(const h16* base, int rowstride, int n0, int kb, int lane) {
  const int hi = (lane >> 4) & 1;
  const h16* p = base + (size_t)(n0 + (lane & 15)) * rowstride + kb + hi * 8;
  v8h lo = *(const v8h*)(p);
  v8h hh = *(const v8h*)(p + 16);
  v16h r;
#pragma unroll
  for (int e = 0; e < 8; ++e) { r[e] = lo[e]; r[e + 8] = hh[e]; }
  return r;
}

__device__ __forceinline__ void flush_u(float* __restrict__ U, int g, int c0w, int lane,
                                        v8f uacc[2][2]) {
  const int hi = (lane >> 4) & 1;
#pragma unroll
  for (int tt = 0; tt < 2; ++tt)
#pragma unroll
    for (int ct = 0; ct < 2; ++ct)
#pragma unroll
      for (int r = 0; r < 8; ++r) {
        const int t = tt * 16 + r + hi * 8;
        const int c = c0w + ct * 16 + (lane & 15);
        const float v = uacc[tt][ct][r];
        if (v != 0.f) atomicAdd(&U[((size_t)g * T_DIM + t) * C_DIM + c], v);
      }
}

// ---------------------------------------------------------------------------
// K0: per-star prep. One block per star (128 blocks x 256 threads).
// q/k/v of stars, A[g,t,:], b0, self-loop score, init m, sumex; zero U.
// ---------------------------------------------------------------------------
__global__ void __launch_bounds__(256) k0_prep(
    const float* __restrict__ stars,
    const float* __restrict__ Wq_w, const float* __restrict__ Wq_b,
    const float* __restrict__ Wk_w, const float* __restrict__ Wk_b,
    const float* __restrict__ Wv_w, const float* __restrict__ Wv_b,
    float* __restrict__ A, float* __restrict__ b0,
    float* __restrict__ scoreself, float* __restrict__ mseg,
    float* __restrict__ sumex, float* __restrict__ vstar,
    float* __restrict__ U)
{
  __shared__ float xs[C_DIM], qs[C_DIM], ks[C_DIM];
  const int s = blockIdx.x;
  const int g = s >> 2, i = s & 3;
  const int c = threadIdx.x;
  xs[c] = stars[s * C_DIM + c];
  __syncthreads();
  float q = Wq_b[c], kk = Wk_b[c], vv = Wv_b[c];
  for (int cc = 0; cc < C_DIM; ++cc) {
    const float x = xs[cc];
    q  += Wq_w[c * C_DIM + cc] * x;
    kk += Wk_w[c * C_DIM + cc] * x;
    vv += Wv_w[c * C_DIM + cc] * x;
  }
  qs[c] = q; ks[c] = kk;
  vstar[s * C_DIM + c] = vv;
  __syncthreads();
#pragma unroll
  for (int h = 0; h < HEADS; ++h) {
    float a = 0.f;
    for (int d = 0; d < 32; ++d) a += qs[h * 32 + d] * Wk_w[(h * 32 + d) * C_DIM + c];
    A[((size_t)g * T_DIM + i * HEADS + h) * C_DIM + c] = a * RSQRT_DH;
  }
  float p1 = qs[c] * Wk_b[c];   // q·bk contribution (per head)
  float p2 = qs[c] * ks[c];     // self-loop score (per head)
  for (int o = 16; o > 0; o >>= 1) { p1 += __shfl_xor(p1, o, 32); p2 += __shfl_xor(p2, o, 32); }
  if ((c & 31) == 0) {
    const int h = c >> 5;
    const int tg = g * T_DIM + i * HEADS + h;
    b0[tg] = p1 * RSQRT_DH;
    const float sc = p2 * RSQRT_DH;
    scoreself[tg] = sc;
    mseg[tg] = sc;           // self loop always present -> safe max seed
    sumex[tg] = 0.f;         // node-only sum of exp
  }
  for (int idx = blockIdx.x * 256 + threadIdx.x; idx < G_NUM * T_DIM * C_DIM; idx += gridDim.x * 256)
    U[idx] = 0.f;
}

// ---------------------------------------------------------------------------
// K1: segment-max pass. 32-node chunks async-staged into LDS (f32), scores
// via WMMA (f32->f16 conversion at fragment gather), atomic max into mseg.
// mseg is only a softmax *shift* (exactly cancels in K2/K3), so it does not
// need to match K2's f16-staged scores bit-for-bit.
// ---------------------------------------------------------------------------
__global__ void __launch_bounds__(256) k1_segmax(
    const float* __restrict__ nodes, const int* __restrict__ batch,
    const float* __restrict__ A, const float* __restrict__ b0,
    float* __restrict__ mseg, int N, int cpb, int nchunks)
{
  __shared__ __align__(16) float xf32[32][C_DIM];  // async-staged node chunk (32 KB)
  __shared__ __align__(16) h16 As[T_DIM][C_DIM];   // A for current graph, [t][k]
  __shared__ float b0s[T_DIM];
  const int tid = threadIdx.x;
  const int lane = tid & 31;
  const int w = tid >> 5;
  int gcur = -1;
  const int chb = blockIdx.x * cpb;
  const int che = min(chb + cpb, nchunks);
  for (int ch = chb; ch < che; ++ch) {
    const int base = ch * 32;
    const int g0 = batch[base];
    const int g1 = batch[min(base + 31, N - 1)];
    const bool mixed = (g0 != g1);                 // block-uniform
    const bool gchange = (!mixed) && (g0 != gcur); // block-uniform
    __syncthreads();
    // async-stage node chunk: 2048 x 16B, 8 issues per thread
    for (int it = tid; it < 32 * 64; it += 256) {
      const int r = it >> 6, q4 = it & 63;
      const int node = base + r;
      float* dst = &xf32[r][q4 * 4];
      if (node < N) {
        async_copy_b128((unsigned int)(size_t)dst, nodes + (size_t)node * C_DIM + q4 * 4);
      } else {
        dst[0] = 0.f; dst[1] = 0.f; dst[2] = 0.f; dst[3] = 0.f;
      }
    }
    if (gchange) {
      for (int it = tid; it < T_DIM * 64; it += 256) {
        const int r = it >> 6, q4 = it & 63;
        float4 av = ((const float4*)(A + ((size_t)g0 * T_DIM + r) * C_DIM))[q4];
        h16* dst = &As[r][q4 * 4];
        dst[0] = (h16)av.x; dst[1] = (h16)av.y; dst[2] = (h16)av.z; dst[3] = (h16)av.w;
      }
      if (tid < T_DIM) b0s[tid] = b0[g0 * T_DIM + tid];
      gcur = g0;
    }
    wait_asynccnt0();          // drain this wave's async copies
    __syncthreads();           // all waves' chunk data visible in LDS
    if (mixed) {
      // rare graph-boundary chunk: scalar f32 path
      const int r = tid >> 3;
      const int node = base + r;
      if (node < N) {
        const int g = batch[node];
        for (int tt = (tid & 7) * 4; tt < (tid & 7) * 4 + 4; ++tt) {
          float sacc = b0[g * T_DIM + tt];
          const float* arow = A + ((size_t)g * T_DIM + tt) * C_DIM;
          for (int cc = 0; cc < C_DIM; ++cc) sacc += arow[cc] * xf32[r][cc];
          atomicMaxF32(&mseg[g * T_DIM + tt], sacc);
        }
      }
    } else if (w < 4) {  // wave-uniform: EXEC all-ones inside
      const int sm = w & 1;
      const int t0 = (w >> 1) * 16;
      v8f acc = {};
#pragma unroll
      for (int kb = 0; kb < C_DIM; kb += 32) {
        v16h af = frag_a_f32(&xf32[0][0], C_DIM, sm * 16, kb, lane);
        v16h bf = frag_b(&As[0][0], C_DIM, t0, kb, lane);
        acc = __builtin_amdgcn_wmma_f32_16x16x32_f16(false, af, false, bf, (short)0, acc, false, false);
      }
      const int t = t0 + (lane & 15);
      const int hi = (lane >> 4) & 1;
      const float bb = b0s[t];              // hoisted: row-invariant
      const int nvalid = N - (base + sm * 16 + hi * 8);  // valid rows in this strip
      float vmax = -3.0e38f;
#pragma unroll
      for (int r = 0; r < 8; ++r)
        if (r < nvalid) vmax = fmaxf(vmax, acc[r] + bb);
      vmax = fmaxf(vmax, __shfl_xor(vmax, 16, 32));
      if (lane < 16) atomicMaxF32(&mseg[gcur * T_DIM + t], vmax);
    }
  }
}

// ---------------------------------------------------------------------------
// K2: exp + accumulation pass. Recompute scores (WMMA), ex=exp(s-m),
// sumex += ex (LDS-reduced), U[g,t,:] += ex * x via WMMA outer product.
// ---------------------------------------------------------------------------
__global__ void __launch_bounds__(256) k2_accum(
    const float* __restrict__ nodes, const int* __restrict__ batch,
    const float* __restrict__ A, const float* __restrict__ b0,
    const float* __restrict__ mseg, float* __restrict__ sumex,
    float* __restrict__ U, int N, int cpb, int nchunks)
{
  __shared__ __align__(16) h16 xs[32][C_DIM];     // [node][k]   (score A-frags)
  __shared__ __align__(16) h16 xsT[C_DIM][32];    // [c][node]   (U B-frags)
  __shared__ __align__(16) h16 As[T_DIM][C_DIM];  // [t][k]
  __shared__ __align__(16) h16 exsT[T_DIM][32];   // [t][node]   (U A-frags)
  __shared__ float b0s[T_DIM], ms[T_DIM], sum_lds[T_DIM];
  const int tid = threadIdx.x;
  const int lane = tid & 31;
  const int w = tid >> 5;
  const int c0w = w * 32;       // each of 8 waves owns a 32-wide channel slice
  int gcur = -1;
  v8f uacc[2][2];
#pragma unroll
  for (int a = 0; a < 2; ++a)
#pragma unroll
    for (int b2 = 0; b2 < 2; ++b2)
#pragma unroll
      for (int r = 0; r < 8; ++r) uacc[a][b2][r] = 0.f;
  const int chb = blockIdx.x * cpb;
  const int che = min(chb + cpb, nchunks);
  for (int ch = chb; ch < che; ++ch) {
    const int base = ch * 32;
    const int g0 = batch[base];
    const int g1 = batch[min(base + 31, N - 1)];
    const bool mixed = (g0 != g1);
    const bool gchange = (!mixed) && (g0 != gcur);
    if (gchange && gcur >= 0) {  // new graph: flush per-graph accumulators
      flush_u(U, gcur, c0w, lane, uacc);
#pragma unroll
      for (int a = 0; a < 2; ++a)
#pragma unroll
        for (int b2 = 0; b2 < 2; ++b2)
#pragma unroll
          for (int r = 0; r < 8; ++r) uacc[a][b2][r] = 0.f;
    }
    __syncthreads();
    for (int it = tid; it < 32 * 64; it += 256) {  // stage chunk (+ transposed copy)
      const int r = it >> 6, q4 = it & 63;
      const int node = base + r;
      float4 xv = make_float4(0.f, 0.f, 0.f, 0.f);
      if (node < N) xv = ((const float4*)(nodes + (size_t)node * C_DIM))[q4];
      const h16 h0 = (h16)xv.x, h1 = (h16)xv.y, h2 = (h16)xv.z, h3 = (h16)xv.w;
      h16* dst = &xs[r][q4 * 4];
      dst[0] = h0; dst[1] = h1; dst[2] = h2; dst[3] = h3;
      xsT[q4 * 4 + 0][r] = h0; xsT[q4 * 4 + 1][r] = h1;
      xsT[q4 * 4 + 2][r] = h2; xsT[q4 * 4 + 3][r] = h3;
    }
    if (gchange) {
      for (int it = tid; it < T_DIM * 64; it += 256) {
        const int r = it >> 6, q4 = it & 63;
        float4 av = ((const float4*)(A + ((size_t)g0 * T_DIM + r) * C_DIM))[q4];
        h16* dst = &As[r][q4 * 4];
        dst[0] = (h16)av.x; dst[1] = (h16)av.y; dst[2] = (h16)av.z; dst[3] = (h16)av.w;
      }
      if (tid < T_DIM) {
        b0s[tid] = b0[g0 * T_DIM + tid];
        ms[tid]  = mseg[g0 * T_DIM + tid];
      }
      gcur = g0;
    }
    if (tid < T_DIM) sum_lds[tid] = 0.f;
    if (ch + 1 < che)
      __builtin_prefetch(nodes + (size_t)(base + 32) * C_DIM + tid * 4, 0, 1);
    __syncthreads();
    if (mixed) {
      // rare boundary chunk: scalar path with direct global atomics
      const int r = tid >> 3;
      const int node = base + r;
      if (node < N) {
        const int g = batch[node];
        for (int tt = (tid & 7) * 4; tt < (tid & 7) * 4 + 4; ++tt) {
          float sacc = b0[g * T_DIM + tt];
          const float* arow = A + ((size_t)g * T_DIM + tt) * C_DIM;
          for (int cc = 0; cc < C_DIM; ++cc) sacc += arow[cc] * (float)xs[r][cc];
          const float ex = __expf(sacc - mseg[g * T_DIM + tt]);
          atomicAdd(&sumex[g * T_DIM + tt], ex);
          float* urow = U + ((size_t)g * T_DIM + tt) * C_DIM;
          for (int cc = 0; cc < C_DIM; ++cc) atomicAdd(&urow[cc], ex * (float)xs[r][cc]);
        }
      }
    } else {
      if (w < 4) {  // phase 1: scores -> exp -> exsT (wave-uniform branch)
        const int sm = w & 1;
        const int t0 = (w >> 1) * 16;
        v8f acc = {};
#pragma unroll
        for (int kb = 0; kb < C_DIM; kb += 32) {
          v16h af = frag_a(&xs[0][0], C_DIM, sm * 16, kb, lane);
          v16h bf = frag_b(&As[0][0], C_DIM, t0, kb, lane);
          acc = __builtin_amdgcn_wmma_f32_16x16x32_f16(false, af, false, bf, (short)0, acc, false, false);
        }
        const int t = t0 + (lane & 15);
        const int hi = (lane >> 4) & 1;
        const float shift = b0s[t] - ms[t];   // hoisted: row-invariant
        const int nvalid = N - (base + sm * 16 + hi * 8);
        float psum = 0.f;
#pragma unroll
        for (int r = 0; r < 8; ++r) {
          const int ln = sm * 16 + r + hi * 8;
          float ex = 0.f;
          if (r < nvalid) ex = __expf(acc[r] + shift);
          exsT[t][ln] = (h16)ex;
          psum += ex;
        }
        if (psum != 0.f) atomicAdd(&sum_lds[t], psum);
      }
      __syncthreads();
      // phase 2: U[g, t0..t0+15 / 16..31, c0w..c0w+31] += exsT x xsT (K=32)
      v16h af0 = frag_a(&exsT[0][0], 32, 0, 0, lane);
      v16h af1 = frag_a(&exsT[0][0], 32, 16, 0, lane);
      v16h bf0 = frag_b(&xsT[0][0], 32, c0w, 0, lane);
      v16h bf1 = frag_b(&xsT[0][0], 32, c0w + 16, 0, lane);
      uacc[0][0] = __builtin_amdgcn_wmma_f32_16x16x32_f16(false, af0, false, bf0, (short)0, uacc[0][0], false, false);
      uacc[0][1] = __builtin_amdgcn_wmma_f32_16x16x32_f16(false, af0, false, bf1, (short)0, uacc[0][1], false, false);
      uacc[1][0] = __builtin_amdgcn_wmma_f32_16x16x32_f16(false, af1, false, bf0, (short)0, uacc[1][0], false, false);
      uacc[1][1] = __builtin_amdgcn_wmma_f32_16x16x32_f16(false, af1, false, bf1, (short)0, uacc[1][1], false, false);
      if (tid < T_DIM) {
        const float sv = sum_lds[tid];
        if (sv != 0.f) atomicAdd(&sumex[gcur * T_DIM + tid], sv);
      }
    }
  }
  if (gcur >= 0) flush_u(U, gcur, c0w, lane, uacc);
}

// ---------------------------------------------------------------------------
// K3: out = LN(elu((Wv·U + bv*sumex + ex_self*v_star)/denom) + stars)
// One block per star, thread = output channel.
// ---------------------------------------------------------------------------
__global__ void __launch_bounds__(256) k3_final(
    const float* __restrict__ stars,
    const float* __restrict__ Wv_w, const float* __restrict__ Wv_b,
    const float* __restrict__ gamma, const float* __restrict__ beta,
    const float* __restrict__ U, const float* __restrict__ mseg,
    const float* __restrict__ sumex, const float* __restrict__ scoreself,
    const float* __restrict__ vstar, float* __restrict__ out)
{
  __shared__ float Ul[HEADS][C_DIM];
  __shared__ float red0[C_DIM], red1[C_DIM];
  const int s = blockIdx.x;
  const int g = s >> 2, i = s & 3;
  const int c = threadIdx.x;
  for (int it = c; it < HEADS * C_DIM; it += 256) {
    const int r = it >> 8, cc = it & 255;
    Ul[r][cc] = U[((size_t)g * T_DIM + i * HEADS + r) * C_DIM + cc];
  }
  __syncthreads();
  const int h = c >> 5;
  const int tg = g * T_DIM + i * HEADS + h;
  const float mt = mseg[tg];
  const float sx = sumex[tg];
  const float es = __expf(scoreself[tg] - mt);
  const float denom = sx + es + 1e-16f;
  float dot = 0.f;
  for (int cc = 0; cc < C_DIM; ++cc) dot += Wv_w[c * C_DIM + cc] * Ul[h][cc];
  const float raw = (dot + Wv_b[c] * sx + es * vstar[s * C_DIM + c]) / denom;
  const float el = raw > 0.f ? raw : (__expf(raw) - 1.f);
  const float ns = el + stars[s * C_DIM + c];
  red0[c] = ns; red1[c] = ns * ns;
  __syncthreads();
  for (int o = 128; o > 0; o >>= 1) {
    if (c < o) { red0[c] += red0[c + o]; red1[c] += red1[c + o]; }
    __syncthreads();
  }
  const float mu = red0[0] * (1.f / C_DIM);
  const float var = red1[0] * (1.f / C_DIM) - mu * mu;
  out[s * C_DIM + c] = (ns - mu) * rsqrtf(var + 1e-5f) * gamma[c] + beta[c];
}

// ---------------------------------------------------------------------------
extern "C" void kernel_launch(void* const* d_in, const int* in_sizes, int n_in,
                              void* d_out, int out_size, void* d_ws, size_t ws_size,
                              hipStream_t stream)
{
  (void)n_in; (void)out_size; (void)ws_size;
  const float* stars = (const float*)d_in[0];
  const float* nodes = (const float*)d_in[1];
  const int*   batch = (const int*)d_in[2];
  const float* Wq_w  = (const float*)d_in[3];
  const float* Wq_b  = (const float*)d_in[4];
  const float* Wk_w  = (const float*)d_in[5];
  const float* Wk_b  = (const float*)d_in[6];
  const float* Wv_w  = (const float*)d_in[7];
  const float* Wv_b  = (const float*)d_in[8];
  const float* lng   = (const float*)d_in[9];
  const float* lnb   = (const float*)d_in[10];
  float* out = (float*)d_out;
  const int N = in_sizes[1] / C_DIM;

  // Workspace layout (floats): ~2.15 MB total.
  float* ws = (float*)d_ws;
  size_t off = 0;
  float* A         = ws + off; off += (size_t)G_NUM * T_DIM * C_DIM;   // 262144
  float* b0        = ws + off; off += G_NUM * T_DIM;                   // 1024
  float* scoreself = ws + off; off += G_NUM * T_DIM;
  float* mseg      = ws + off; off += G_NUM * T_DIM;
  float* sumex     = ws + off; off += G_NUM * T_DIM;
  float* vstar     = ws + off; off += (size_t)G_NUM * NUM_STAR * C_DIM; // 32768
  float* U         = ws + off; off += (size_t)G_NUM * T_DIM * C_DIM;    // 262144

  k0_prep<<<G_NUM * NUM_STAR, 256, 0, stream>>>(stars, Wq_w, Wq_b, Wk_w, Wk_b, Wv_w, Wv_b,
                                                A, b0, scoreself, mseg, sumex, vstar, U);
  const int nchunks = (N + 31) / 32;
  const int nb = 256;
  const int cpb = (nchunks + nb - 1) / nb;
  k1_segmax<<<nb, 256, 0, stream>>>(nodes, batch, A, b0, mseg, N, cpb, nchunks);
  k2_accum<<<nb, 256, 0, stream>>>(nodes, batch, A, b0, mseg, sumex, U, N, cpb, nchunks);
  k3_final<<<G_NUM * NUM_STAR, 256, 0, stream>>>(stars, Wv_w, Wv_b, lng, lnb,
                                                 U, mseg, sumex, scoreself, vstar, out);
}
